// SRGNN_55860344652411
// MI455X (gfx1250) — matrix operand order repaired and checked by reading, not compiled, not run on hardware
//
#include <hip/hip_runtime.h>
#include <hip/hip_bf16.h>
#include <math.h>

// ---------------------------------------------------------------------------
// SR-GNN forward on MI455X (gfx1250): all GEMMs via v_wmma_f32_16x16x32_bf16,
// bf16 storage everywhere, f32 accumulate, async LDS staging where possible.
// ---------------------------------------------------------------------------

typedef __attribute__((ext_vector_type(16))) __bf16 bf16x16;
typedef __attribute__((ext_vector_type(8)))  float  f32x8;

#define BB   4096
#define NN   50
#define LL   50
#define HH   128
#define DD   128

#if defined(__AMDGCN__) && __has_builtin(__builtin_amdgcn_global_load_async_to_lds_b128) && \
    __has_builtin(__builtin_amdgcn_s_wait_asynccnt)
#define HAVE_ASYNC_LDS 1
// Builtin prototype (from hipcc diagnostic): v4i addrspace(1)*, v4i addrspace(3)*, Ii, Ii
typedef int v4i __attribute__((vector_size(16)));
typedef __attribute__((address_space(1))) v4i* gv4i_p;
typedef __attribute__((address_space(3))) v4i* lv4i_p;
#endif

__device__ __forceinline__ f32x8 wmma_bf16(bf16x16 a, bf16x16 b, f32x8 c) {
  return __builtin_amdgcn_wmma_f32_16x16x32_bf16(false, a, false, b, (short)0, c,
                                                 false, false);
}

// A-fragment (16x32 bf16) from a row-major LDS tile, row stride ld (padded).
// ISA 7.12.2: lanes 0-15 hold row m, K={k0..k0+7, k0+16..k0+23};
// lanes 16-31: K={k0+8..k0+15, k0+24..k0+31}. Two 16B contiguous loads.
__device__ __forceinline__ bf16x16 load_afrag(const __bf16* a, int ld, int k0, int lane) {
  int m = lane & 15, hi = lane >> 4;
  const __bf16* p = a + m * ld + k0 + hi * 8;
  bf16x16 r;
#pragma unroll
  for (int j = 0; j < 8; ++j) { r[j] = p[j]; r[8 + j] = p[16 + j]; }
  return r;
}

// B-fragment (32x16 bf16) = X^T where X is row-major bf16 [Nout, K] (global
// weight or transposed LDS tile). Lane n reads K={k0+8h..+7, k0+16+8h..+7}.
__device__ __forceinline__ bf16x16 load_bfrag_bf(const __bf16* X, int ld, int nbase,
                                                 int k0, int lane) {
  const __bf16* p = X + (long)(nbase + (lane & 15)) * ld + k0 + (lane >> 4) * 8;
  bf16x16 r;
#pragma unroll
  for (int t = 0; t < 8; ++t) { r[t] = p[t]; r[8 + t] = p[16 + t]; }
  return r;
}

// Stage a ROWS x KD bf16 tile (contiguous rows, row stride lda) into padded LDS
// (row stride LDP). Uses CDNA5 async global->LDS copies when available.
template <int ROWS, int KD, int LDP>
__device__ __forceinline__ void stage_tile(__bf16* dst, const __bf16* src, long row0,
                                           int lda, int tid) {
#ifdef HAVE_ASYNC_LDS
  constexpr int CHUNKS = ROWS * KD / 8;  // 16-byte chunks
  constexpr int PER = CHUNKS / 256;
#pragma unroll
  for (int j = 0; j < PER; ++j) {
    int c = tid * PER + j;
    int r = c / (KD / 8), kc = c % (KD / 8);
    const __bf16* g = src + (row0 + r) * (long)lda + kc * 8;
    __bf16* l = dst + r * LDP + kc * 8;
    __builtin_amdgcn_global_load_async_to_lds_b128((gv4i_p)(void*)g, (lv4i_p)(void*)l,
                                                   0, 0);
  }
#else
  constexpr int PER = ROWS * KD / 256;
#pragma unroll
  for (int j = 0; j < PER; ++j) {
    int e = tid * PER + j;
    int r = e / KD, cc = e % KD;
    dst[r * LDP + cc] = src[(row0 + r) * (long)lda + cc];
  }
#endif
}

__device__ __forceinline__ void stage_wait() {
#ifdef HAVE_ASYNC_LDS
  __builtin_amdgcn_s_wait_asynccnt(0);
#endif
  __syncthreads();
}

// ---------------------------------------------------------------------------
// Weight f32 -> bf16 pre-conversion.
// ---------------------------------------------------------------------------
__global__ __launch_bounds__(256) void cvt_bf16(const float* __restrict__ src,
                                                __bf16* __restrict__ dst, int n) {
  int i = blockIdx.x * 256 + threadIdx.x;
  if (i < n) dst[i] = (__bf16)src[i];
}

// ---------------------------------------------------------------------------
// Generic GEMM: out[M,128] = A[M,KDIM] @ W[128,KDIM]^T + bias.
// 32 rows per workgroup (2 row-tiles/wave share each B-fragment),
// 8 waves each own a 16-column tile. W pre-converted bf16.
// ---------------------------------------------------------------------------
template <int KDIM, bool A_F32, bool GATHER>
__global__ __launch_bounds__(256) void gemm128(const void* __restrict__ Asrc,
                                               const int* __restrict__ rowidx, int lda,
                                               const __bf16* __restrict__ Wb,
                                               const float* __restrict__ bias,
                                               float* __restrict__ outF,
                                               __bf16* __restrict__ outB) {
  constexpr int LDP = KDIM + 8;
  __shared__ __bf16 As[32 * LDP];
  long m0 = (long)blockIdx.x * 32;
  int tid = threadIdx.x;
  if constexpr (!A_F32 && !GATHER) {
    stage_tile<32, KDIM, LDP>(As, (const __bf16*)Asrc, m0, lda, tid);
  } else {
    constexpr int PER = (32 * KDIM) / 256;
#pragma unroll
    for (int j = 0; j < PER; ++j) {
      int e = tid * PER + j;
      int r = e / KDIM, c = e % KDIM;
      long sr = GATHER ? (long)rowidx[m0 + r] : (m0 + r);
      float v = A_F32 ? ((const float*)Asrc)[sr * (long)lda + c]
                      : (float)((const __bf16*)Asrc)[sr * (long)lda + c];
      As[r * LDP + c] = (__bf16)v;
    }
  }
  stage_wait();
  int wave = tid >> 5, lane = tid & 31;
  int c0 = wave * 16;
  f32x8 acc0 = {}, acc1 = {};
#pragma unroll
  for (int k0 = 0; k0 < KDIM; k0 += 32) {
    bf16x16 b = load_bfrag_bf(Wb, KDIM, c0, k0, lane);
    acc0 = wmma_bf16(load_afrag(As, LDP, k0, lane), b, acc0);
    acc1 = wmma_bf16(load_afrag(As + 16 * LDP, LDP, k0, lane), b, acc1);
  }
  int col = c0 + (lane & 15), hi = lane >> 4;
  float bv = bias ? bias[col] : 0.f;
#pragma unroll
  for (int i = 0; i < 8; ++i) {
    long r0 = m0 + i + 8 * hi, r1 = r0 + 16;
    float v0 = acc0[i] + bv, v1 = acc1[i] + bv;
    if (outF) { outF[r0 * 128 + col] = v0; outF[r1 * 128 + col] = v1; }
    if (outB) { outB[r0 * 128 + col] = (__bf16)v0; outB[r1 * 128 + col] = (__bf16)v1; }
  }
}

// ---------------------------------------------------------------------------
// Batched adjacency GEMM: per batch b,
//   a_in  = A[b][:, :50] @ hv_in[b] + b_iah ; a_out = A[b][:, 50:] @ hv_out[b] + b_oah
// hv tiles staged TRANSPOSED so B-fragments are contiguous ds_load_b128 pairs.
// av stores [a_in | a_out] bf16 [B*N, 256].
// ---------------------------------------------------------------------------
__global__ __launch_bounds__(256) void adj_gemm(const float* __restrict__ A,
                                                const __bf16* __restrict__ hvin,
                                                const __bf16* __restrict__ hvout,
                                                const float* __restrict__ b_iah,
                                                const float* __restrict__ b_oah,
                                                __bf16* __restrict__ av) {
  __shared__ __bf16 Ain[64 * 72];
  __shared__ __bf16 Aout[64 * 72];
  __shared__ __bf16 HinT[128 * 72];   // HinT[n][k] = hv_in[b, k, n]
  __shared__ __bf16 HoutT[128 * 72];
  int b = blockIdx.x, tid = threadIdx.x;
#pragma unroll
  for (int j = 0; j < 16; ++j) {
    int e = tid * 16 + j;
    int i = e >> 6, c = e & 63;
    float vi = 0.f, vo = 0.f;
    if (i < NN && c < NN) {
      const float* row = A + ((long)b * NN + i) * (2 * NN);
      vi = row[c];
      vo = row[NN + c];
    }
    Ain[i * 72 + c] = (__bf16)vi;
    Aout[i * 72 + c] = (__bf16)vo;
  }
#pragma unroll
  for (int j = 0; j < 32; ++j) {
    int e = tid * 32 + j;  // k = e>>7 in [0,64), n = e&127 (coalesced global reads)
    int k = e >> 7, n = e & 127;
    __bf16 vi = (__bf16)0.f, vo = (__bf16)0.f;
    if (k < NN) {
      long off = ((long)b * NN + k) * 128 + n;
      vi = hvin[off];
      vo = hvout[off];
    }
    HinT[n * 72 + k] = vi;
    HoutT[n * 72 + k] = vo;
  }
  __syncthreads();
  int wave = tid >> 5, lane = tid & 31;
  int c0 = wave * 16;
  f32x8 ai[4] = {}, ao[4] = {};
#pragma unroll
  for (int k0 = 0; k0 < 64; k0 += 32) {
    bf16x16 b1 = load_bfrag_bf(HinT, 72, c0, k0, lane);
    bf16x16 b2 = load_bfrag_bf(HoutT, 72, c0, k0, lane);
#pragma unroll
    for (int rt = 0; rt < 4; ++rt) {
      ai[rt] = wmma_bf16(load_afrag(Ain + rt * 16 * 72, 72, k0, lane), b1, ai[rt]);
      ao[rt] = wmma_bf16(load_afrag(Aout + rt * 16 * 72, 72, k0, lane), b2, ao[rt]);
    }
  }
  int col = c0 + (lane & 15), hi = lane >> 4;
  float bi = b_iah[col], bo = b_oah[col];
#pragma unroll
  for (int rt = 0; rt < 4; ++rt)
#pragma unroll
    for (int i = 0; i < 8; ++i) {
      int rrow = rt * 16 + i + 8 * hi;
      if (rrow < NN) {
        long base = ((long)b * NN + rrow) * 256;
        av[base + col] = (__bf16)(ai[rt][i] + bi);
        av[base + 128 + col] = (__bf16)(ao[rt][i] + bo);
      }
    }
}

// ---------------------------------------------------------------------------
// Fused GRU: gi = av @ w_ih^T + b_ih (K=256->384), gh = hid0 @ w_hh^T + b_hh,
// gates in registers, new hidden bf16. 32 rows/WG; each wave owns columns
// c0, c0+128, c0+256 (r/z/n aligned per lane); B-fragments feed 2 row-tiles.
// ---------------------------------------------------------------------------
__global__ __launch_bounds__(256) void gru_fused(const __bf16* __restrict__ av,
                                                 const __bf16* __restrict__ hid0,
                                                 const __bf16* __restrict__ w_ih,
                                                 const float* __restrict__ b_ih,
                                                 const __bf16* __restrict__ w_hh,
                                                 const float* __restrict__ b_hh,
                                                 __bf16* __restrict__ hid1) {
  __shared__ __bf16 Xs[32 * 264];
  __shared__ __bf16 Hs[32 * 136];
  long m0 = (long)blockIdx.x * 32;
  int tid = threadIdx.x;
  stage_tile<32, 256, 264>(Xs, av, m0, 256, tid);
  stage_tile<32, 128, 136>(Hs, hid0, m0, 128, tid);
  stage_wait();
  int wave = tid >> 5, lane = tid & 31;
  int c0 = wave * 16, col = c0 + (lane & 15), hi = lane >> 4;
  f32x8 gi[2][3] = {};
  f32x8 gh[2][3] = {};
#pragma unroll
  for (int k0 = 0; k0 < 256; k0 += 32) {
    bf16x16 a0 = load_afrag(Xs, 264, k0, lane);
    bf16x16 a1 = load_afrag(Xs + 16 * 264, 264, k0, lane);
#pragma unroll
    for (int g = 0; g < 3; ++g) {
      bf16x16 b = load_bfrag_bf(w_ih, 256, c0 + 128 * g, k0, lane);
      gi[0][g] = wmma_bf16(a0, b, gi[0][g]);
      gi[1][g] = wmma_bf16(a1, b, gi[1][g]);
    }
  }
#pragma unroll
  for (int k0 = 0; k0 < 128; k0 += 32) {
    bf16x16 a0 = load_afrag(Hs, 136, k0, lane);
    bf16x16 a1 = load_afrag(Hs + 16 * 136, 136, k0, lane);
#pragma unroll
    for (int g = 0; g < 3; ++g) {
      bf16x16 b = load_bfrag_bf(w_hh, 128, c0 + 128 * g, k0, lane);
      gh[0][g] = wmma_bf16(a0, b, gh[0][g]);
      gh[1][g] = wmma_bf16(a1, b, gh[1][g]);
    }
  }
  float bir = b_ih[col], biz = b_ih[col + 128], binn = b_ih[col + 256];
  float bhr = b_hh[col], bhz = b_hh[col + 128], bhn = b_hh[col + 256];
#pragma unroll
  for (int mt = 0; mt < 2; ++mt)
#pragma unroll
    for (int i = 0; i < 8; ++i) {
      int r = mt * 16 + i + 8 * hi;
      float rg = 1.f / (1.f + __expf(-((gi[mt][0][i] + bir) + (gh[mt][0][i] + bhr))));
      float zg = 1.f / (1.f + __expf(-((gi[mt][1][i] + biz) + (gh[mt][1][i] + bhz))));
      float ng = tanhf((gi[mt][2][i] + binn) + rg * (gh[mt][2][i] + bhn));
      float hold = (float)Hs[r * 136 + col];
      hid1[(m0 + r) * 128 + col] = (__bf16)(ng + zg * (hold - ng));
    }
}

// ---------------------------------------------------------------------------
// ht[b] = hid1[b, alias[b, sum(mask[b]) - 1]]
// ---------------------------------------------------------------------------
__global__ __launch_bounds__(128) void gather_ht(const int* __restrict__ mask,
                                                 const int* __restrict__ alias,
                                                 const __bf16* __restrict__ hid1,
                                                 __bf16* __restrict__ ht) {
  __shared__ int lastb;
  int b = blockIdx.x;
  if (threadIdx.x == 0) {
    int s = 0;
    for (int l = 0; l < LL; ++l) s += mask[b * LL + l];
    lastb = s - 1;
  }
  __syncthreads();
  int node = alias[b * LL + lastb];
  ht[(long)b * 128 + threadIdx.x] = hid1[((long)b * NN + node) * 128 + threadIdx.x];
}

// ---------------------------------------------------------------------------
// Attention + masked weighted sum, one workgroup per batch.
// ---------------------------------------------------------------------------
__global__ __launch_bounds__(256) void attention(const int* __restrict__ alias,
                                                 const int* __restrict__ mask,
                                                 const __bf16* __restrict__ hid1,
                                                 const float* __restrict__ q1,
                                                 const __bf16* __restrict__ W2b,
                                                 const float* __restrict__ b2,
                                                 const float* __restrict__ W3,
                                                 float* __restrict__ aout) {
  __shared__ __bf16 Ss[64 * 136];
  __shared__ float alpha[64];
  int b = blockIdx.x, tid = threadIdx.x;
#pragma unroll
  for (int j = 0; j < 32; ++j) {
    int e = tid * 32 + j;
    int l = e >> 7, c = e & 127;
    __bf16 v = (__bf16)0.f;
    if (l < LL) {
      int node = alias[b * LL + l];
      v = hid1[((long)b * NN + node) * 128 + c];
    }
    Ss[l * 136 + c] = v;
  }
  if (tid < 64) alpha[tid] = 0.f;
  __syncthreads();
  int wave = tid >> 5, lane = tid & 31;
  int c0 = wave * 16, col = c0 + (lane & 15), hi = lane >> 4;
  float q1v = q1[(long)b * 128 + col] + b2[col];
  float w3v = W3[col];
  f32x8 acc[4] = {};
#pragma unroll
  for (int k0 = 0; k0 < 128; k0 += 32) {
    bf16x16 bb = load_bfrag_bf(W2b, 128, c0, k0, lane);
#pragma unroll
    for (int rt = 0; rt < 4; ++rt)
      acc[rt] = wmma_bf16(load_afrag(Ss + rt * 16 * 136, 136, k0, lane), bb, acc[rt]);
  }
#pragma unroll
  for (int rt = 0; rt < 4; ++rt)
#pragma unroll
    for (int i = 0; i < 8; ++i) {
      float s = 1.f / (1.f + __expf(-(acc[rt][i] + q1v)));
      atomicAdd(&alpha[rt * 16 + i + 8 * hi], s * w3v);
    }
  __syncthreads();
  if (tid < 128) {
    float a = 0.f;
    for (int l = 0; l < LL; ++l)
      a += alpha[l] * (float)mask[b * LL + l] * (float)Ss[l * 136 + tid];
    aout[(long)b * 128 + tid] = a;
  }
}

// ---------------------------------------------------------------------------
// Final: av2 = [a | ht] @ Wt^T + bt ; bvec = emb[item_indices] @ W_et^T + b_et;
// terms[b] = y*log(sig(dot)) + (1-y)*log1p(-sig(dot)). 16 batches / WG.
// ---------------------------------------------------------------------------
__global__ __launch_bounds__(256) void final_score(const float* __restrict__ aout,
                                                   const __bf16* __restrict__ ht,
                                                   const __bf16* __restrict__ Wtb,
                                                   const float* __restrict__ bt,
                                                   const int* __restrict__ item_indices,
                                                   const float* __restrict__ emb,
                                                   const __bf16* __restrict__ Wetb,
                                                   const float* __restrict__ b_et,
                                                   const float* __restrict__ labels,
                                                   float* __restrict__ terms) {
  __shared__ __bf16 Acat[16 * 264];
  __shared__ __bf16 Ev[16 * 136];
  __shared__ float dots[16];
  int b0 = blockIdx.x * 16, tid = threadIdx.x;
#pragma unroll
  for (int j = 0; j < 16; ++j) {
    int e = tid * 16 + j;
    int r = e >> 8, c = e & 255;
    __bf16 v = (c < 128) ? (__bf16)aout[(long)(b0 + r) * 128 + c]
                         : ht[(long)(b0 + r) * 128 + (c - 128)];
    Acat[r * 264 + c] = v;
  }
#pragma unroll
  for (int j = 0; j < 8; ++j) {
    int e = tid * 8 + j;
    int r = e >> 7, c = e & 127;
    Ev[r * 136 + c] = (__bf16)emb[(long)item_indices[b0 + r] * 128 + c];
  }
  if (tid < 16) dots[tid] = 0.f;
  __syncthreads();
  int wave = tid >> 5, lane = tid & 31;
  int c0 = wave * 16, col = c0 + (lane & 15), hi = lane >> 4;
  f32x8 av2 = {}, bv = {};
#pragma unroll
  for (int k0 = 0; k0 < 256; k0 += 32)
    av2 = wmma_bf16(load_afrag(Acat, 264, k0, lane),
                    load_bfrag_bf(Wtb, 256, c0, k0, lane), av2);
#pragma unroll
  for (int k0 = 0; k0 < 128; k0 += 32)
    bv = wmma_bf16(load_afrag(Ev, 136, k0, lane),
                   load_bfrag_bf(Wetb, 128, c0, k0, lane), bv);
  float btv = bt[col], bev = b_et[col];
#pragma unroll
  for (int i = 0; i < 8; ++i)
    atomicAdd(&dots[i + 8 * hi], (av2[i] + btv) * (bv[i] + bev));
  __syncthreads();
  if (tid < 16) {
    float s = 1.f / (1.f + __expf(-dots[tid]));
    const float eps = 1e-7f;
    s = fminf(fmaxf(s, eps), 1.f - eps);
    float y = labels[b0 + tid];
    terms[b0 + tid] = y * __logf(s) + (1.f - y) * log1pf(-s);
  }
}

__global__ __launch_bounds__(256) void reduce_loss(const float* __restrict__ terms,
                                                   float* __restrict__ out) {
  __shared__ float sm[256];
  float s = 0.f;
  for (int i = threadIdx.x; i < BB; i += 256) s += terms[i];
  sm[threadIdx.x] = s;
  __syncthreads();
  for (int st = 128; st > 0; st >>= 1) {
    if (threadIdx.x < st) sm[threadIdx.x] += sm[threadIdx.x + st];
    __syncthreads();
  }
  if (threadIdx.x == 0) out[0] = -sm[0] / (float)BB;
}

// ---------------------------------------------------------------------------
extern "C" void kernel_launch(void* const* d_in, const int* in_sizes, int n_in,
                              void* d_out, int out_size, void* d_ws, size_t ws_size,
                              hipStream_t stream) {
  const float* A            = (const float*)d_in[0];
  const int*   mask         = (const int*)d_in[1];
  const float* labels       = (const float*)d_in[2];
  const int*   alias        = (const int*)d_in[3];
  const int*   items        = (const int*)d_in[4];
  const int*   item_indices = (const int*)d_in[5];
  const float* emb          = (const float*)d_in[6];
  const float* W_et = (const float*)d_in[7];  const float* b_et = (const float*)d_in[8];
  const float* W_in = (const float*)d_in[9];  const float* b_in = (const float*)d_in[10];
  const float* W_out= (const float*)d_in[11]; const float* b_out= (const float*)d_in[12];
  const float* b_iah= (const float*)d_in[13]; const float* b_oah= (const float*)d_in[14];
  const float* w_ih = (const float*)d_in[15]; const float* b_ih = (const float*)d_in[16];
  const float* w_hh = (const float*)d_in[17]; const float* b_hh = (const float*)d_in[18];
  const float* W1   = (const float*)d_in[19]; const float* b1   = (const float*)d_in[20];
  const float* W2   = (const float*)d_in[21]; const float* b2   = (const float*)d_in[22];
  const float* W3   = (const float*)d_in[23];
  const float* Wt   = (const float*)d_in[24]; const float* bt   = (const float*)d_in[25];

  const long BN = (long)BB * NN;  // 204800
  char* ws = (char*)d_ws;
  size_t off = 0;
  auto carve = [&](size_t bytes) {
    size_t o = off;
    off += (bytes + 255) & ~(size_t)255;
    return o;
  };
  __bf16* hid0   = (__bf16*)(ws + carve(BN * 128 * 2));  // 52.4 MB
  size_t  r1     = carve(BN * 128 * 2);                  // hv_in, reused as hid1
  size_t  r2     = carve(BN * 128 * 2);                  // hv_out, reused as small bufs
  __bf16* av     = (__bf16*)(ws + carve(BN * 256 * 2));  // 104.8 MB
  // bf16 weight shadows (~0.5 MB)
  __bf16* Wet_b  = (__bf16*)(ws + carve(HH * DD * 2));
  __bf16* Win_b  = (__bf16*)(ws + carve(HH * HH * 2));
  __bf16* Wout_b = (__bf16*)(ws + carve(HH * HH * 2));
  __bf16* wih_b  = (__bf16*)(ws + carve(3 * HH * 2 * HH * 2));
  __bf16* whh_b  = (__bf16*)(ws + carve(3 * HH * HH * 2));
  __bf16* W1_b   = (__bf16*)(ws + carve(HH * HH * 2));
  __bf16* W2_b   = (__bf16*)(ws + carve(HH * HH * 2));
  __bf16* Wt_b   = (__bf16*)(ws + carve(HH * 2 * HH * 2));

  __bf16* hv_in  = (__bf16*)(ws + r1);
  __bf16* hv_out = (__bf16*)(ws + r2);
  __bf16* hid1   = hv_in;  // hv_in dead after adj_gemm
  // small buffers carved out of the (dead after adj_gemm) hv_out region
  __bf16* ht    = (__bf16*)(ws + r2);              // 1 MB
  float*  q1    = (float*)(ws + r2 + (2l << 20));  // 2 MB
  float*  aoutb = (float*)(ws + r2 + (6l << 20));  // 2 MB
  float*  terms = (float*)(ws + r2 + (10l << 20)); // 16 KB

  const int blk = 256;
  auto cvt = [&](const float* s, __bf16* d, int n) {
    cvt_bf16<<<(n + 255) / 256, 256, 0, stream>>>(s, d, n);
  };
  cvt(W_et, Wet_b, HH * DD);
  cvt(W_in, Win_b, HH * HH);
  cvt(W_out, Wout_b, HH * HH);
  cvt(w_ih, wih_b, 3 * HH * 2 * HH);
  cvt(w_hh, whh_b, 3 * HH * HH);
  cvt(W1, W1_b, HH * HH);
  cvt(W2, W2_b, HH * HH);
  cvt(Wt, Wt_b, HH * 2 * HH);

  // 1) hidden0 = emb[items] @ W_et^T + b_et  (gathered f32 A)
  gemm128<128, true, true><<<BN / 32, blk, 0, stream>>>(emb, items, DD, Wet_b, b_et,
                                                        nullptr, hid0);
  // 2) hv_in/hv_out
  gemm128<128, false, false><<<BN / 32, blk, 0, stream>>>(hid0, nullptr, HH, Win_b, b_in,
                                                          nullptr, hv_in);
  gemm128<128, false, false><<<BN / 32, blk, 0, stream>>>(hid0, nullptr, HH, Wout_b, b_out,
                                                          nullptr, hv_out);
  // 3) batched adjacency GEMMs -> av = [a_in | a_out]
  adj_gemm<<<BB, blk, 0, stream>>>(A, hv_in, hv_out, b_iah, b_oah, av);
  // 4) fused GRU -> hid1
  gru_fused<<<BN / 32, blk, 0, stream>>>(av, hid0, wih_b, b_ih, whh_b, b_hh, hid1);
  // 5) ht gather, q1 = ht @ W1^T + b1
  gather_ht<<<BB, 128, 0, stream>>>(mask, alias, hid1, ht);
  gemm128<128, false, false><<<BB / 32, blk, 0, stream>>>(ht, nullptr, HH, W1_b, b1,
                                                          q1, nullptr);
  // 6) attention -> aout
  attention<<<BB, blk, 0, stream>>>(alias, mask, hid1, q1, W2_b, b2, W3, aoutb);
  // 7) final score + per-batch BCE terms, 8) reduce
  final_score<<<BB / 16, blk, 0, stream>>>(aoutb, ht, Wt_b, bt, item_indices, emb, Wet_b,
                                           b_et, labels, terms);
  reduce_loss<<<1, blk, 0, stream>>>(terms, (float*)d_out);
  (void)in_sizes; (void)n_in; (void)out_size; (void)ws_size;
}
